// ScanNet_2482491097353
// MI455X (gfx1250) — compile-verified, matrix operand order checked by reading, not run on hardware
//
#include <hip/hip_runtime.h>
#include <math.h>

#define B_    2
#define L_    1024
#define LA_   8192
#define KDIM_ 276      // 4 * (1 + 68) effective K of the big GEMM

typedef float    v2f  __attribute__((ext_vector_type(2)));
typedef float    v8f  __attribute__((ext_vector_type(8)));
typedef _Float16 v16h __attribute__((ext_vector_type(16)));

// ---------------------------------------------------------------- zero init
__global__ void zero_f32(float* __restrict__ p, int n) {
  int t = blockIdx.x * blockDim.x + threadIdx.x;
  if (t < n) p[t] = 0.f;
}

// ------------------------------------------------ emb_at = emb_atom[attr_atom]
__global__ void embed_atom(const int* __restrict__ attr_atom,   // (B, LA)
                           const float* __restrict__ emb_atom,  // (13, 12)
                           float* __restrict__ out) {           // (B, LA, 12)
  int t = blockIdx.x * blockDim.x + threadIdx.x;                // B*LA*12 threads
  int f = t % 12;
  int i = t / 12;                                               // b*LA + ia
  int a = attr_atom[i];
  out[t] = emb_atom[a * 12 + f];
}

// -------------------------------------------------- KNN (top-4) + gather
// One query per thread; candidate coords staged through LDS as float4 so the
// hot loop issues a single ds_load_b128 broadcast per candidate (issue-bound
// on wave32). Strict '<' with ascending index scan reproduces jax.lax.top_k's
// stable lowest-index-first tie-breaking on equal distances.
template <int NPTS, int NATTR>
__global__ void knn_gather(const float* __restrict__ coords,  // (B, NPTS, 3)
                           const float* __restrict__ attrs,   // (B, NPTS, NATTR)
                           float* __restrict__ out) {         // (B, NPTS, 4, 1+NATTR)
  __shared__ float4 sc[256];
  const int tid = threadIdx.x;
  const int gq  = blockIdx.x * 256 + tid;       // NPTS % 256 == 0 -> same batch per block
  const int b   = gq / NPTS;
  const int q   = gq - b * NPTS;
  const float* cb = coords + (size_t)b * NPTS * 3;
  const float qx = cb[q * 3 + 0], qy = cb[q * 3 + 1], qz = cb[q * 3 + 2];

  float bd0 = 3.4e38f, bd1 = 3.4e38f, bd2 = 3.4e38f, bd3 = 3.4e38f;
  int   bi0 = 0, bi1 = 0, bi2 = 0, bi3 = 0;

  for (int t0 = 0; t0 < NPTS; t0 += 256) {
    __syncthreads();
    const int j = t0 + tid;
    sc[tid] = make_float4(cb[j * 3 + 0], cb[j * 3 + 1], cb[j * 3 + 2], 0.f);
    __syncthreads();
    #pragma unroll 8
    for (int jj = 0; jj < 256; ++jj) {
      const float4 c = sc[jj];                  // one ds_load_b128, broadcast
      const float dx = qx - c.x, dy = qy - c.y, dz = qz - c.z;
      const float d2 = dx * dx + dy * dy + dz * dz;
      if (d2 < bd3) {
        const int idx = t0 + jj;
        if (d2 < bd0) {
          bd3 = bd2; bi3 = bi2; bd2 = bd1; bi2 = bi1; bd1 = bd0; bi1 = bi0;
          bd0 = d2;  bi0 = idx;
        } else if (d2 < bd1) {
          bd3 = bd2; bi3 = bi2; bd2 = bd1; bi2 = bi1;
          bd1 = d2;  bi1 = idx;
        } else if (d2 < bd2) {
          bd3 = bd2; bi3 = bi2;
          bd2 = d2;  bi2 = idx;
        } else {
          bd3 = d2;  bi3 = idx;
        }
      }
    }
  }

  const float bd[4] = {bd0, bd1, bd2, bd3};
  const int   bi[4] = {bi0, bi1, bi2, bi3};
  const float* ab = attrs + (size_t)b * NPTS * NATTR;
  float* op = out + (size_t)(b * NPTS + q) * (4 * (1 + NATTR));
  #pragma unroll
  for (int kk = 0; kk < 4; ++kk) {
    const float d2 = bd[kk];
    op[kk * (1 + NATTR)] = (d2 > 0.f) ? sqrtf(d2) : 0.f;   // matches reference where()
    const float* ap = ab + (size_t)bi[kk] * NATTR;
    for (int f = 0; f < NATTR; ++f) op[kk * (1 + NATTR) + 1 + f] = ap[f];
  }
}

// ------------------------------------- segment_sum via global FP32 atomics
__global__ void seg_sum(const float* __restrict__ atom_feats,   // (B, LA, 52)
                        const int* __restrict__ indices_atom,   // (B, LA, 1)
                        float* __restrict__ emb_full) {         // (B, L, 68), cols 16..67
  int t = blockIdx.x * blockDim.x + threadIdx.x;                // B*LA*52 threads
  int f = t % 52;
  int i = t / 52;                                               // b*LA + ia
  int b = i / LA_;
  int seg = indices_atom[i];
  atomicAdd(&emb_full[(size_t)(b * L_ + seg) * 68 + 16 + f], atom_feats[t]);
}

// ------------------------------------- emb_full[:, :, 0:16] = attr_aa @ W_aa + b_aa
__global__ void embed_aa(const float* __restrict__ attr_aa,  // (B, L, 20)
                         const float* __restrict__ W_aa,     // (20, 16)
                         const float* __restrict__ b_aa,     // (16,)
                         float* __restrict__ emb_full) {     // (B, L, 68)
  int t = blockIdx.x * blockDim.x + threadIdx.x;             // B*L threads
  const float* a = attr_aa + (size_t)t * 20;
  float* o = emb_full + (size_t)t * 68;
  #pragma unroll
  for (int oo = 0; oo < 16; ++oo) {
    float s = b_aa[oo];
    for (int c = 0; c < 20; ++c) s = fmaf(a[c], W_aa[c * 16 + oo], s);
    o[oo] = s;
  }
}

// --------------------------- H = aa_feats(2048x276) @ W1[0:276,0:256] + b1
// 256 threads = 8 waves/block; block owns a 16-row M tile; each wave owns two
// 16-col N tiles. K loop = 69 x V_WMMA_F32_16X16X4_F32 per tile (f32 path).
__global__ void gemm_wmma(const float* __restrict__ A,     // (2048, 276)
                          const float* __restrict__ W1,    // (6032, 256); rows >=276 unused (zero pad)
                          const float* __restrict__ bias,  // (256,)
                          float* __restrict__ H) {         // (2048, 256)
  const int lane = threadIdx.x & 31;
  const int wave = threadIdx.x >> 5;
  const int l16  = lane & 15;
  const int half = lane >> 4;                 // 0: K lo pair, 1: K hi pair
  const int rowBase = blockIdx.x * 16;
  const int n0 = wave * 32 + l16;
  const int n1 = n0 + 16;
  const float* Arow = A + (size_t)(rowBase + l16) * KDIM_;

  v8f acc0 = {};
  v8f acc1 = {};

#if defined(__has_builtin) && __has_builtin(__builtin_amdgcn_wmma_f32_16x16x4_f32)
  // f32 A 16x4 layout: lanes 0-15 hold K=0,1; lanes 16-31 hold K=2,3 (per row).
  // B 4x16: rows striped across lanes within a VGPR, lo/hi halves split rows.
  for (int k0 = 0; k0 < KDIM_; k0 += 4) {
    const int ka = k0 + 2 * half;                       // 8-byte aligned (ka even)
    const v2f a = *(const v2f*)(Arow + ka);
    v2f b0;  b0.x  = W1[(size_t)ka * 256 + n0];  b0.y  = W1[(size_t)(ka + 1) * 256 + n0];
    v2f b1v; b1v.x = W1[(size_t)ka * 256 + n1];  b1v.y = W1[(size_t)(ka + 1) * 256 + n1];
    acc0 = __builtin_amdgcn_wmma_f32_16x16x4_f32(false, a, false, b0,  (short)0, acc0, false, false);
    acc1 = __builtin_amdgcn_wmma_f32_16x16x4_f32(false, a, false, b1v, (short)0, acc1, false, false);
  }
#else
  // Fallback: f16 WMMA 16x16x32 with on-the-fly conversion (K padded to 288).
  for (int k0 = 0; k0 < 288; k0 += 32) {
    v16h af, bf0, bf1;
    #pragma unroll
    for (int v = 0; v < 8; ++v) {
      const int kb = ((v < 4) ? (2 * v) : (16 + 2 * (v - 4))) + half * 8;
      const int k  = k0 + kb;
      const float a0 = (k     < KDIM_) ? Arow[k]     : 0.f;
      const float a1 = (k + 1 < KDIM_) ? Arow[k + 1] : 0.f;
      af[2 * v]     = (_Float16)a0;
      af[2 * v + 1] = (_Float16)a1;
      const float w00 = (k     < KDIM_) ? W1[(size_t)k * 256 + n0]       : 0.f;
      const float w01 = (k + 1 < KDIM_) ? W1[(size_t)(k + 1) * 256 + n0] : 0.f;
      bf0[2 * v]     = (_Float16)w00;
      bf0[2 * v + 1] = (_Float16)w01;
      const float w10 = (k     < KDIM_) ? W1[(size_t)k * 256 + n1]       : 0.f;
      const float w11 = (k + 1 < KDIM_) ? W1[(size_t)(k + 1) * 256 + n1] : 0.f;
      bf1[2 * v]     = (_Float16)w10;
      bf1[2 * v + 1] = (_Float16)w11;
    }
    acc0 = __builtin_amdgcn_wmma_f32_16x16x32_f16(false, af, false, bf0, (short)0, acc0, false, false);
    acc1 = __builtin_amdgcn_wmma_f32_16x16x32_f16(false, af, false, bf1, (short)0, acc1, false, false);
  }
#endif

  // 16x16 f32 C/D layout: VGPR v: lanes 0-15 -> M=v, lanes 16-31 -> M=v+8; N=lane%16.
  #pragma unroll
  for (int v = 0; v < 8; ++v) {
    const int r = rowBase + v + 8 * half;
    H[(size_t)r * 256 + n0] = acc0[v] + bias[n0];
    H[(size_t)r * 256 + n1] = acc1[v] + bias[n1];
  }
}

// ----------------------- LayerNorm(256) + exact GELU + 256->1 head per row
__global__ void ln_gelu_head(const float* __restrict__ H,     // (2048, 256)
                             const float* __restrict__ g,     // ln_g
                             const float* __restrict__ be,    // ln_b
                             const float* __restrict__ W2,    // (256,1)
                             const float* __restrict__ b2,    // (1,)
                             float* __restrict__ out) {       // (2048,)
  __shared__ float red[256];
  const int row = blockIdx.x, t = threadIdx.x;
  const float h = H[(size_t)row * 256 + t];

  red[t] = h; __syncthreads();
  for (int s = 128; s > 0; s >>= 1) { if (t < s) red[t] += red[t + s]; __syncthreads(); }
  const float mu = red[0] * (1.f / 256.f);
  __syncthreads();

  const float c = h - mu;
  red[t] = c * c; __syncthreads();
  for (int s = 128; s > 0; s >>= 1) { if (t < s) red[t] += red[t + s]; __syncthreads(); }
  const float var = red[0] * (1.f / 256.f);
  __syncthreads();

  const float x  = c * rsqrtf(var + 1e-5f) * g[t] + be[t];
  const float ge = 0.5f * x * (1.f + erff(x * 0.7071067811865475f));   // exact GELU
  red[t] = ge * W2[t]; __syncthreads();
  for (int s = 128; s > 0; s >>= 1) { if (t < s) red[t] += red[t + s]; __syncthreads(); }
  if (t == 0) out[row] = red[0] + b2[0];
}

// ---------------------------------------------------------------- launcher
extern "C" void kernel_launch(void* const* d_in, const int* in_sizes, int n_in,
                              void* d_out, int out_size, void* d_ws, size_t ws_size,
                              hipStream_t stream) {
  (void)in_sizes; (void)n_in; (void)out_size; (void)ws_size;

  const float* coord_aa     = (const float*)d_in[0];
  const float* attr_aa      = (const float*)d_in[1];
  const float* coord_atom   = (const float*)d_in[4];
  const int*   attr_atom    = (const int*)d_in[5];
  const int*   indices_atom = (const int*)d_in[7];
  const float* W_aa         = (const float*)d_in[8];
  const float* b_aa         = (const float*)d_in[9];
  const float* emb_atom     = (const float*)d_in[10];
  const float* W1           = (const float*)d_in[11];
  const float* b1           = (const float*)d_in[12];
  const float* ln_g         = (const float*)d_in[13];
  const float* ln_b         = (const float*)d_in[14];
  const float* W2           = (const float*)d_in[15];
  const float* b2           = (const float*)d_in[16];

  float* ws         = (float*)d_ws;
  float* emb_at     = ws;                 // B*LA*12  = 196608
  float* atom_feats = ws + 196608;        // B*LA*52  = 851968
  float* emb_full   = ws + 1048576;       // B*L*68   = 139264
  float* aa_feats   = ws + 1187840;       // B*L*276  = 565248
  float* H          = ws + 1753088;       // 2048*256 = 524288  (total ~8.7 MB)
  float* logits     = (float*)d_out;      // (2, 1024)

  zero_f32   <<<544, 256, 0, stream>>>(emb_full, B_ * L_ * 68);
  embed_atom <<<768, 256, 0, stream>>>(attr_atom, emb_atom, emb_at);
  knn_gather<LA_, 12><<<64, 256, 0, stream>>>(coord_atom, emb_at, atom_feats);
  seg_sum    <<<3328, 256, 0, stream>>>(atom_feats, indices_atom, emb_full);
  embed_aa   <<<8, 256, 0, stream>>>(attr_aa, W_aa, b_aa, emb_full);
  knn_gather<L_, 68><<<8, 256, 0, stream>>>(coord_aa, emb_full, aa_feats);
  gemm_wmma  <<<128, 256, 0, stream>>>(aa_feats, W1, b1, H);
  ln_gelu_head<<<2048, 256, 0, stream>>>(H, ln_g, ln_b, W2, b2, logits);
}
